// SpatioTemporelLSTM_v2_11458972745755
// MI455X (gfx1250) — matrix operand order repaired and checked by reading, not compiled, not run on hardware
//
#include <hip/hip_runtime.h>
#include <math.h>

// ---------------- types for WMMA ----------------
typedef __attribute__((ext_vector_type(16))) __bf16       v16bf;
typedef __attribute__((ext_vector_type(8)))  float        v8f;
typedef __attribute__((ext_vector_type(4)))  unsigned int v4u;

union Frag16 { v4u q[2]; v16bf v; };

#define HDIM 256
#define TENC 96
#define TDEC 16

// ---------------- bf16 helpers ----------------
__device__ __forceinline__ unsigned short f2bf(float f) {
    unsigned int u = __float_as_uint(f);
    u += 0x7fffu + ((u >> 16) & 1u);     // round-to-nearest-even
    return (unsigned short)(u >> 16);
}
__device__ __forceinline__ float bf2f(unsigned short h) {
    return __uint_as_float(((unsigned int)h) << 16);
}
__device__ __forceinline__ float sigmf(float x) { return 1.0f / (1.0f + expf(-x)); }

// ---------------- WMMA fragment loads ----------------
// A: 16x32 bf16 tile from row-major [16][ldk] ushort buffer.
// lane L: row = L&15 ; elems 0..7 -> K = k0 + 8*(L>=16) + {0..7}
//                       elems 8..15 -> K = +16 of that.  (ISA 7.12.2)
__device__ __forceinline__ v16bf load_a_frag(const unsigned short* base, int ldk, int k0, int lane) {
    int row = lane & 15;
    int kb  = k0 + ((lane >> 4) << 3);
    const unsigned short* p = base + row * ldk + kb;
    Frag16 f;
    f.q[0] = *(const v4u*)(p);
    f.q[1] = *(const v4u*)(p + 16);
    return f.v;
}
// B: 32x16 bf16 tile of W^T where W is row-major [N][ldk] ushort.
// lane L: col = n0 + (L&15) ; elems 0..7 -> K = k0 + 8*(L>=16) + {0..7}, elems 8..15 -> K = +16 of that
__device__ __forceinline__ v16bf load_b_frag(const unsigned short* w, int ldk, int n0, int k0, int lane) {
    int col = n0 + (lane & 15);
    int kb  = k0 + ((lane >> 4) << 3);
    const unsigned short* p = w + col * ldk + kb;
    Frag16 f;
    f.q[0] = *(const v4u*)(p);
    f.q[1] = *(const v4u*)(p + 16);
    return f.v;
}
#define WMMA_BF16(a, b, c) \
    __builtin_amdgcn_wmma_f32_16x16x32_bf16(false, (a), false, (b), (short)0, (c), false, false)

// ---------------- kernel 1: weight repack to bf16 ----------------
// Wg[1024][512] = [Wih_e | Whh_e] ; also Wi_s, We_s, Wd_t bf16 copies.
__global__ __launch_bounds__(256) void prep_weights(
    const float* __restrict__ WihE, const float* __restrict__ WhhE,
    const float* __restrict__ WiS,  const float* __restrict__ WeS,
    const float* __restrict__ WdT,
    unsigned short* __restrict__ Wg_bf,  unsigned short* __restrict__ WiS_bf,
    unsigned short* __restrict__ WeS_bf, unsigned short* __restrict__ WdT_bf)
{
    int idx = blockIdx.x * 256 + threadIdx.x;       // grid covers 1024*512
    if (idx < 1024 * 512) {
        int n = idx >> 9, k = idx & 511;
        float v = (k < 256) ? WihE[n * 256 + k] : WhhE[n * 256 + (k - 256)];
        Wg_bf[idx] = f2bf(v);
    }
    if (idx < 256 * 256) WiS_bf[idx] = f2bf(WiS[idx]);
    if (idx < 256 * 512) WeS_bf[idx] = f2bf(WeS[idx]);
    if (idx < 256 * 256) WdT_bf[idx] = f2bf(WdT[idx]);
}

// ---------------- kernel 2: assemble encoder inputs (bf16) ----------------
// inputs[b,t,:] = [p_q(235) | cfg(10) | E_hour(5) | E_wd(3) | E_season(3)]
__global__ __launch_bounds__(256) void prep_inputs(
    const float* __restrict__ pq, const float* __restrict__ cfgin,
    const int* __restrict__ timein,
    const float* __restrict__ Wcfg, const float* __restrict__ bcfg,
    const float* __restrict__ Eh, const float* __restrict__ Ew, const float* __restrict__ Es,
    unsigned short* __restrict__ inputs_bf)
{
    int row = blockIdx.x;                 // b*96 + t
    int col = threadIdx.x;
    float v;
    if (col < 235) {
        v = pq[(size_t)row * 235 + col];
    } else if (col < 245) {
        int j = col - 235;
        float s = bcfg[j];
        const float* c = cfgin + (size_t)row * 32;
        #pragma unroll
        for (int k = 0; k < 32; ++k) s += c[k] * Wcfg[j * 32 + k];
        v = s;
    } else if (col < 250) {
        v = Eh[timein[row * 3 + 0] * 5 + (col - 245)];
    } else if (col < 253) {
        v = Ew[timein[row * 3 + 1] * 3 + (col - 250)];
    } else {
        v = Es[timein[row * 3 + 2] * 3 + (col - 253)];
    }
    inputs_bf[(size_t)row * HDIM + col] = f2bf(v);
}

// ---------------- kernel 3: generic Y = A@W^T + b, K=N=256, WMMA ----------------
// A: bf16 [R][256], W: bf16 [256][256], Y: bf16 [R][256]
__global__ __launch_bounds__(256) void gemm256_bf16(
    const unsigned short* __restrict__ A, const unsigned short* __restrict__ W,
    const float* __restrict__ bias, unsigned short* __restrict__ Y)
{
    int tid = threadIdx.x, lane = tid & 31, wave = tid >> 5;
    int stripe = blockIdx.x >> 1, half = blockIdx.x & 1;
    int n0 = half * 128 + wave * 16;
    const unsigned short* Ab = A + (size_t)stripe * 16 * HDIM;
    v8f acc = {};
    for (int kk = 0; kk < 8; ++kk) {
        int k0 = kk * 32;
        v16bf a = load_a_frag(Ab, HDIM, k0, lane);
        v16bf b = load_b_frag(W, HDIM, n0, k0, lane);
        acc = WMMA_BF16(a, b, acc);
    }
    int col = n0 + (lane & 15);
    int rowoff = (lane >> 4) * 8;
    float bc = bias[col];
    #pragma unroll
    for (int r = 0; r < 8; ++r) {
        int row = stripe * 16 + rowoff + r;
        Y[(size_t)row * HDIM + col] = f2bf(acc[r] + bc);
    }
}

// ---------------- kernel 4: persistent encoder ----------------
// One WG per 16-row batch tile; 8 waves split 256 hidden cols (32 each).
// Per step: attention WMMA (K=512,N=256), gates WMMA (K=512,N=1024), LSTM update.
__global__ __launch_bounds__(256) void encoder_kernel(
    const unsigned short* __restrict__ inputs_bf,  // [B][96][256]
    const unsigned short* __restrict__ wi_bf,      // [B][96][256]
    const unsigned short* __restrict__ WeS_bf,     // [256][512]
    const unsigned short* __restrict__ Wg_bf,      // [1024][512]
    const float* __restrict__ VdS, const float* __restrict__ bVdS,
    const float* __restrict__ bih, const float* __restrict__ bhh,
    unsigned short* __restrict__ mid_bf)           // [B][96][256]
{
    __shared__ unsigned short hc[16][512];   // cols 0..255 = h, 256..511 = c (bf16)
    __shared__ unsigned short xp[16][256];   // x_t * a (bf16)
    __shared__ float a_part[8][16];
    __shared__ float a_val[16];

    const int tid = threadIdx.x, lane = tid & 31, wave = tid >> 5;
    const int rbase = blockIdx.x * 16;
    const int colA = lane & 15;
    const int rowoff = (lane >> 4) * 8;

    for (int i = tid; i < 16 * 512; i += 256) (&hc[0][0])[i] = 0;
    v8f cfr[2] = {};   // per-wave fp32 cell state fragments (cols 32*wave..+31)
    __syncthreads();

    for (int step = 0; step < TENC; ++step) {
        // ---- attention score: s = tanh([h|c]@We_s^T + wi_t); a = s@Vd_s^T + bVd_s
        v8f sacc[2] = {};
        for (int kk = 0; kk < 16; ++kk) {
            int k0 = kk * 32;
            v16bf a = load_a_frag(&hc[0][0], 512, k0, lane);
            #pragma unroll
            for (int t2 = 0; t2 < 2; ++t2) {
                v16bf b = load_b_frag(WeS_bf, 512, wave * 32 + t2 * 16, k0, lane);
                sacc[t2] = WMMA_BF16(a, b, sacc[t2]);
            }
        }
        float p[8];
        #pragma unroll
        for (int r = 0; r < 8; ++r) p[r] = 0.0f;
        #pragma unroll
        for (int t2 = 0; t2 < 2; ++t2) {
            int col = wave * 32 + t2 * 16 + colA;
            float vd = VdS[col];
            #pragma unroll
            for (int r = 0; r < 8; ++r) {
                int row = rowoff + r;
                float w = bf2f(wi_bf[((size_t)(rbase + row) * TENC + step) * HDIM + col]);
                p[r] += tanhf(sacc[t2][r] + w) * vd;
            }
        }
        #pragma unroll
        for (int m = 1; m < 16; m <<= 1) {
            #pragma unroll
            for (int r = 0; r < 8; ++r) p[r] += __shfl_xor(p[r], m, 32);
        }
        if (colA == 0) {
            #pragma unroll
            for (int r = 0; r < 8; ++r) a_part[wave][rowoff + r] = p[r];
        }
        __syncthreads();
        if (tid < 16) {
            float s = bVdS[0];
            #pragma unroll
            for (int w8 = 0; w8 < 8; ++w8) s += a_part[w8][tid];
            a_val[tid] = s;
        }
        __syncthreads();
        // ---- x' = x_t * a (bf16 into LDS)
        for (int row = 0; row < 16; ++row) {
            float x = bf2f(inputs_bf[((size_t)(rbase + row) * TENC + step) * HDIM + tid]);
            xp[row][tid] = f2bf(x * a_val[row]);
        }
        __syncthreads();
        // ---- gates = [x'|h] @ [Wih|Whh]^T   (each wave: 4 gates x 2 col tiles)
        v8f g[4][2] = {};
        for (int kk = 0; kk < 16; ++kk) {
            int k0 = kk * 32;
            v16bf a = (k0 < 256) ? load_a_frag(&xp[0][0], 256, k0, lane)
                                 : load_a_frag(&hc[0][0], 512, k0 - 256, lane);
            #pragma unroll
            for (int G = 0; G < 4; ++G) {
                #pragma unroll
                for (int t2 = 0; t2 < 2; ++t2) {
                    v16bf b = load_b_frag(Wg_bf, 512, G * 256 + wave * 32 + t2 * 16, k0, lane);
                    g[G][t2] = WMMA_BF16(a, b, g[G][t2]);
                }
            }
        }
        __syncthreads();   // all reads of hc done before rewrite
        // ---- LSTM elementwise update (c stays in registers)
        #pragma unroll
        for (int t2 = 0; t2 < 2; ++t2) {
            int col = wave * 32 + t2 * 16 + colA;
            float bI = bih[col]       + bhh[col];
            float bF = bih[256 + col] + bhh[256 + col];
            float bG = bih[512 + col] + bhh[512 + col];
            float bO = bih[768 + col] + bhh[768 + col];
            #pragma unroll
            for (int r = 0; r < 8; ++r) {
                int row = rowoff + r;
                float iv = sigmf(g[0][t2][r] + bI);
                float fv = sigmf(g[1][t2][r] + bF);
                float gv = tanhf(g[2][t2][r] + bG);
                float ov = sigmf(g[3][t2][r] + bO);
                float c2 = fv * cfr[t2][r] + iv * gv;
                float h2 = ov * tanhf(c2);
                cfr[t2][r] = c2;
                unsigned short h2b = f2bf(h2);
                hc[row][col] = h2b;
                hc[row][256 + col] = f2bf(c2);
                mid_bf[((size_t)(rbase + row) * TENC + step) * HDIM + col] = h2b;
            }
        }
        __syncthreads();
    }
}

// ---------------- kernel 5: persistent decoder (hidden size 1) ----------------
__global__ __launch_bounds__(256) void decoder_kernel(
    const unsigned short* __restrict__ mid_bf,   // [B][96][256]
    const unsigned short* __restrict__ wdm_bf,   // [B][96][256]
    const float* __restrict__ Wd2, const float* __restrict__ VdT, const float* __restrict__ bVdT,
    const float* __restrict__ WihD, const float* __restrict__ WhhD,
    const float* __restrict__ bihD, const float* __restrict__ bhhD,
    float* __restrict__ out)                     // [B][16]
{
    __shared__ float q[16][256];
    __shared__ float att[16][96];
    __shared__ float ctx[16][256];
    __shared__ float hd[16], cd[16], pv[16];
    __shared__ float gbuf[16][4];

    const int tid = threadIdx.x, lane = tid & 31, wave = tid >> 5;
    const int rbase = blockIdx.x * 16;

    if (tid < 16) {
        hd[tid] = 0.0f; cd[tid] = 0.0f;
        pv[tid] = bf2f(mid_bf[((size_t)(rbase + tid) * TENC + 95) * HDIM + 2]); // mid[:,-1,2]
    }
    __syncthreads();

    for (int step = 0; step < TDEC; ++step) {
        // q = [h|c] @ Wd2_t^T   (Wd2_t is (256,2))
        {
            float w0 = Wd2[tid * 2 + 0], w1 = Wd2[tid * 2 + 1];
            for (int row = 0; row < 16; ++row) q[row][tid] = hd[row] * w0 + cd[row] * w1;
        }
        __syncthreads();
        // att[row][t] = tanh(q[row]+wd_mid[row][t]) . Vd_t + bVd_t  (wave-cooperative dot)
        for (int pidx = wave; pidx < 16 * TENC; pidx += 8) {
            int row = pidx / TENC, t = pidx % TENC;
            const unsigned short* wp = wdm_bf + ((size_t)(rbase + row) * TENC + t) * HDIM;
            float s = 0.0f;
            for (int j = lane; j < HDIM; j += 32)
                s += tanhf(q[row][j] + bf2f(wp[j])) * VdT[j];
            #pragma unroll
            for (int m = 1; m < 32; m <<= 1) s += __shfl_xor(s, m, 32);
            if (lane == 0) att[row][t] = s + bVdT[0];
        }
        __syncthreads();
        // ctx[row][:] = sum_t att[row][t] * mid[row][t][:]
        {
            for (int row = 0; row < 16; ++row) {
                const unsigned short* mp = mid_bf + (size_t)(rbase + row) * TENC * HDIM + tid;
                float s = 0.0f;
                for (int t = 0; t < TENC; ++t) s += att[row][t] * bf2f(mp[(size_t)t * HDIM]);
                ctx[row][tid] = s;
            }
        }
        __syncthreads();
        // tiny LSTM gates: g[k] = Wih_d[k,:256].ctx + Wih_d[k,256]*prev + bih + Whh_d[k]*h + bhh
        for (int d = wave; d < 64; d += 8) {
            int row = d >> 2, k = d & 3;
            const float* wr = WihD + k * 257;
            float s = 0.0f;
            for (int j = lane; j < HDIM; j += 32) s += wr[j] * ctx[row][j];
            #pragma unroll
            for (int m = 1; m < 32; m <<= 1) s += __shfl_xor(s, m, 32);
            if (lane == 0)
                gbuf[row][k] = s + wr[256] * pv[row] + bihD[k] + WhhD[k] * hd[row] + bhhD[k];
        }
        __syncthreads();
        if (tid < 16) {
            float iv = sigmf(gbuf[tid][0]);
            float fv = sigmf(gbuf[tid][1]);
            float gv = tanhf(gbuf[tid][2]);
            float ov = sigmf(gbuf[tid][3]);
            float c2 = fv * cd[tid] + iv * gv;
            float h2 = ov * tanhf(c2);
            cd[tid] = c2; hd[tid] = h2; pv[tid] = h2;
            out[(size_t)(rbase + tid) * TDEC + step] = h2;
        }
        __syncthreads();
    }
}

// ---------------- host launcher ----------------
extern "C" void kernel_launch(void* const* d_in, const int* in_sizes, int n_in,
                              void* d_out, int out_size, void* d_ws, size_t ws_size,
                              hipStream_t stream) {
    const float* in_pq   = (const float*)d_in[0];
    const float* in_cfg  = (const float*)d_in[1];
    const int*   in_time = (const int*)  d_in[2];
    const float* W_cfg   = (const float*)d_in[3];
    const float* b_cfg   = (const float*)d_in[4];
    const float* E_hour  = (const float*)d_in[5];
    const float* E_wd    = (const float*)d_in[6];
    const float* E_season= (const float*)d_in[7];
    const float* Wih_e   = (const float*)d_in[8];
    const float* Whh_e   = (const float*)d_in[9];
    const float* bih_e   = (const float*)d_in[10];
    const float* bhh_e   = (const float*)d_in[11];
    const float* Wih_d   = (const float*)d_in[12];
    const float* Whh_d   = (const float*)d_in[13];
    const float* bih_d   = (const float*)d_in[14];
    const float* bhh_d   = (const float*)d_in[15];
    const float* Wi_s    = (const float*)d_in[16];
    const float* bi_s    = (const float*)d_in[17];
    const float* We_s    = (const float*)d_in[18];
    const float* Vd_s    = (const float*)d_in[19];
    const float* bVd_s   = (const float*)d_in[20];
    const float* Wd_t    = (const float*)d_in[21];
    const float* bd_t    = (const float*)d_in[22];
    const float* Wd2_t   = (const float*)d_in[23];
    const float* Vd_t    = (const float*)d_in[24];
    const float* bVd_t   = (const float*)d_in[25];
    float* out = (float*)d_out;

    const int B = in_sizes[0] / (TENC * 235);   // 1024
    const int R = B * TENC;                      // 98304 rows

    // workspace carve-up (all 16B-aligned)
    unsigned char* ws = (unsigned char*)d_ws;
    size_t o = 0;
    const size_t big = (size_t)R * HDIM * sizeof(unsigned short);   // 48 MiB
    unsigned short* inputs_bf = (unsigned short*)(ws + o); o += big;
    unsigned short* wi_bf     = (unsigned short*)(ws + o); o += big;
    unsigned short* mid_bf    = (unsigned short*)(ws + o); o += big;
    unsigned short* wdm_bf    = (unsigned short*)(ws + o); o += big;
    unsigned short* Wg_bf     = (unsigned short*)(ws + o); o += (size_t)1024 * 512 * 2;
    unsigned short* WeS_bf    = (unsigned short*)(ws + o); o += (size_t)256 * 512 * 2;
    unsigned short* WiS_bf    = (unsigned short*)(ws + o); o += (size_t)256 * 256 * 2;
    unsigned short* WdT_bf    = (unsigned short*)(ws + o); o += (size_t)256 * 256 * 2;

    // 1) weights -> bf16
    prep_weights<<<(1024 * 512) / 256, 256, 0, stream>>>(
        Wih_e, Whh_e, Wi_s, We_s, Wd_t, Wg_bf, WiS_bf, WeS_bf, WdT_bf);
    // 2) assemble inputs
    prep_inputs<<<R, 256, 0, stream>>>(
        in_pq, in_cfg, in_time, W_cfg, b_cfg, E_hour, E_wd, E_season, inputs_bf);
    // 3) wi = inputs @ Wi_s^T + bi_s
    gemm256_bf16<<<(R / 16) * 2, 256, 0, stream>>>(inputs_bf, WiS_bf, bi_s, wi_bf);
    // 4) encoder recurrence (persistent WGs)
    encoder_kernel<<<B / 16, 256, 0, stream>>>(
        inputs_bf, wi_bf, WeS_bf, Wg_bf, Vd_s, bVd_s, bih_e, bhh_e, mid_bf);
    // 5) wd_mid = mid @ Wd_t^T + bd_t
    gemm256_bf16<<<(R / 16) * 2, 256, 0, stream>>>(mid_bf, WdT_bf, bd_t, wdm_bf);
    // 6) decoder
    decoder_kernel<<<B / 16, 256, 0, stream>>>(
        mid_bf, wdm_bf, Wd2_t, Vd_t, bVd_t, Wih_d, Whh_d, bih_d, bhh_d, out);
}
